// Attention_70635032150531
// MI455X (gfx1250) — matrix operand order repaired and checked by reading, not compiled, not run on hardware
//
#include <hip/hip_runtime.h>
#include <hip/hip_bf16.h>
#include <cstdint>

// ---------------------------------------------------------------------------
// Types for CDNA5 WMMA (wave32): V_WMMA_F32_16X16X32_BF16
// ---------------------------------------------------------------------------
typedef __attribute__((ext_vector_type(16))) __bf16         v16bf;
typedef __attribute__((ext_vector_type(8)))  float          f32x8;
typedef __attribute__((ext_vector_type(8)))  unsigned short u16x8;
typedef __attribute__((ext_vector_type(16))) unsigned short u16x16;

#define HEADS    16
#define DIM_HEAD 64
#define SEQ      2048
#define BATCH    4
#define DMODEL   1024
#define ROWS     (BATCH * SEQ)        // 8192
#define QKV_N    (3 * DMODEL)         // 3072
#define ATTN_SCALE 0.125f             // 64^-0.5

// fp32 -> bf16 round-to-nearest-even (bit twiddle; no __bf16 arithmetic needed)
__device__ __forceinline__ unsigned short f2bf(float f) {
  unsigned u = __builtin_bit_cast(unsigned, f);
  u += 0x7FFFu + ((u >> 16) & 1u);
  return (unsigned short)(u >> 16);
}

__device__ __forceinline__ u16x16 cat8(u16x8 lo, u16x8 hi) {
  u16x16 r;
#pragma unroll
  for (int i = 0; i < 8; ++i) { r[i] = lo[i]; r[i + 8] = hi[i]; }
  return r;
}

__device__ __forceinline__ f32x8 wmma_bf16(u16x16 a, u16x16 b, f32x8 c) {
  // 8 args: (neg_a, A, neg_b, B, c_mod, C, reuse_a, reuse_b)
  return __builtin_amdgcn_wmma_f32_16x16x32_bf16(
      false, __builtin_bit_cast(v16bf, a),
      false, __builtin_bit_cast(v16bf, b),
      (short)0, c, false, false);
}

__device__ __forceinline__ f32x8 zero8() {
  f32x8 z;
#pragma unroll
  for (int i = 0; i < 8; ++i) z[i] = 0.0f;
  return z;
}

// xor-shuffle across lanes of the wave32 (ds_bpermute; wave32 uses idx[6:2])
__device__ __forceinline__ float xor_f(float v, int mask) {
  int idx = (((int)(threadIdx.x & 31) ^ mask) << 2);
  return __builtin_bit_cast(
      float, __builtin_amdgcn_ds_bpermute(idx, __builtin_bit_cast(int, v)));
}

// ---------------------------------------------------------------------------
// Async global->LDS staging (CDNA5 GLOBAL_LOAD_ASYNC_TO_LDS_B128, ASYNCcnt)
// Generic __shared__ pointers are {aperture, lds_offset}; truncating to 32
// bits yields the wave-relative LDS byte address the instruction expects.
// ---------------------------------------------------------------------------
__device__ __forceinline__ void async_b128_to_lds(unsigned short* lds_dst,
                                                  const unsigned short* g) {
  unsigned loff = (unsigned)(uintptr_t)lds_dst;
  unsigned long long ga = (unsigned long long)(uintptr_t)g;
  asm volatile("global_load_async_to_lds_b128 %0, %1, off"
               :: "v"(loff), "v"(ga) : "memory");
}

__device__ __forceinline__ void wait_async0() {
#if __has_builtin(__builtin_amdgcn_s_wait_asynccnt)
  __builtin_amdgcn_s_wait_asynccnt(0);
#else
  asm volatile("s_wait_asynccnt 0" ::: "memory");
#endif
}

// ---------------------------------------------------------------------------
// fp32 -> bf16 converters (straight + transposing, for N-major weight layout)
// ---------------------------------------------------------------------------
__global__ void cvt_kernel(const float* __restrict__ src,
                           unsigned short* __restrict__ dst, int n) {
  for (int i = blockIdx.x * blockDim.x + threadIdx.x; i < n;
       i += gridDim.x * blockDim.x)
    dst[i] = f2bf(src[i]);
}

// src[R][C] (row-major) -> dst[C][R]; dims compile-time so idx math is cheap
template <int R, int C>
__global__ void cvt_tr_kernel(const float* __restrict__ src,
                              unsigned short* __restrict__ dst) {
  for (int i = blockIdx.x * blockDim.x + threadIdx.x; i < R * C;
       i += gridDim.x * blockDim.x) {
    int r = i / C, c = i - r * C;
    dst[(size_t)c * R + r] = f2bf(src[i]);
  }
}

// ---------------------------------------------------------------------------
// Issue one block-wide async fill of a 64-col x 32-k bf16 B chunk into LDS.
// 256 threads x b128 = 4KB = whole chunk. ldsB layout: [buf][col_local][32].
// ---------------------------------------------------------------------------
__device__ __forceinline__ void fill_b_chunk(const unsigned short* __restrict__ Bt,
                                             unsigned short* ldsB, int buf,
                                             int col0, int k0, int t) {
  int cl = t >> 2, seg = t & 3;                       // 64 cols x 4 b128 segs
  const unsigned short* g = Bt + (size_t)(col0 + cl) * DMODEL + k0 + seg * 8;
  unsigned short* l = ldsB + buf * (64 * 32) + cl * 32 + seg * 8;
  async_b128_to_lds(l, g);
}

// ---------------------------------------------------------------------------
// Core 16x64 per-wave GEMM tile with block-shared, double-buffered B in LDS.
// A row-major [*,1024] bf16; Bt is N-major (column n of B = row n of Bt).
// All 8 waves of the block share col0 -> B fetched once per block per chunk.
// ---------------------------------------------------------------------------
__device__ __forceinline__ void gemm16x64_lds(
    const unsigned short* __restrict__ A,
    const unsigned short* __restrict__ Bt,
    unsigned short* ldsB,              // [2][64][32] bf16
    int row0, int col0, f32x8 acc[4]) {
  int t = threadIdx.x;
  int lane = t & 31, m = lane & 15, hi = lane >> 4;

  fill_b_chunk(Bt, ldsB, 0, col0, 0, t);
  for (int k0 = 0; k0 < DMODEL; k0 += 32) {
    int buf = (k0 >> 5) & 1;
    wait_async0();                     // my fill of `buf` has landed in LDS
    __syncthreads();                   // everyone's fill of `buf` has landed

    const unsigned short* ap = A + (size_t)(row0 + m) * DMODEL + k0 + 8 * hi;
    u16x16 av = cat8(*(const u16x8*)ap, *(const u16x8*)(ap + 16));
    const unsigned short* lb = ldsB + buf * (64 * 32);
#pragma unroll
    for (int nn = 0; nn < 4; ++nn) {
      const unsigned short* bp = lb + (nn * 16 + m) * 32 + 8 * hi;
      u16x16 bv = cat8(*(const u16x8*)bp, *(const u16x8*)(bp + 16));
      acc[nn] = wmma_bf16(av, bv, acc[nn]);
    }

    __syncthreads();                   // all waves done reading buf^1's slot
    if (k0 + 32 < DMODEL) fill_b_chunk(Bt, ldsB, buf ^ 1, col0, k0 + 32, t);
  }
}

// ---------------------------------------------------------------------------
// Kernel 1: qkv = x @ w_qkv ; scatter into Q (scaled), K [b,h,n,d] and
//           V transposed [b,h,d,n] — all bf16 for the attention stage.
// grid (3072/64, 8192/128), block 256 (8 waves; wave w owns 16x64 tile)
// ---------------------------------------------------------------------------
__global__ __launch_bounds__(256) void qkv_gemm_kernel(
    const unsigned short* __restrict__ xb,
    const unsigned short* __restrict__ wqkvt,
    unsigned short* __restrict__ qb,
    unsigned short* __restrict__ kb,
    unsigned short* __restrict__ vt) {
  __shared__ __align__(16) unsigned short ldsB[2 * 64 * 32];
  int lane = threadIdx.x & 31, wid = threadIdx.x >> 5;
  int m = lane & 15, hi = lane >> 4;
  int row0 = blockIdx.y * 128 + wid * 16;
  int col0 = blockIdx.x * 64;

  f32x8 acc[4];
#pragma unroll
  for (int nn = 0; nn < 4; ++nn) acc[nn] = zero8();

  gemm16x64_lds(xb, wqkvt, ldsB, row0, col0, acc);

#pragma unroll
  for (int nn = 0; nn < 4; ++nn) {
    int col = col0 + nn * 16 + m;
    int sec = col >> 10;            // 0=q 1=k 2=v
    int cc  = col & 1023;
    int hh  = cc >> 6, dd = cc & 63;
#pragma unroll
    for (int r = 0; r < 8; ++r) {
      int rg = row0 + r + 8 * hi;
      int bb = rg >> 11, ns = rg & 2047;
      size_t bh = (size_t)bb * HEADS + hh;
      float v = acc[nn][r];
      if (sec == 0)
        qb[(bh * SEQ + ns) * DIM_HEAD + dd] = f2bf(v * ATTN_SCALE);
      else if (sec == 1)
        kb[(bh * SEQ + ns) * DIM_HEAD + dd] = f2bf(v);
      else
        vt[(bh * DIM_HEAD + dd) * SEQ + ns] = f2bf(v);
    }
  }
}

// ---------------------------------------------------------------------------
// Kernel 2: causal flash attention. One wave (block of 32) per 16-row Q tile
// per (b,h). 32-key tiles; S via 4 WMMAs, online softmax in fp32, P bounced
// through LDS (C-layout -> A-layout transpose), O via 4 WMMAs per tile.
// Single-wave blocks -> causal trip-count divergence needs no barriers.
// grid (64, 128), block 32.
// ---------------------------------------------------------------------------
__global__ __launch_bounds__(32) void attn_kernel(
    const unsigned short* __restrict__ qb,
    const unsigned short* __restrict__ kb,
    const unsigned short* __restrict__ vt,
    unsigned short* __restrict__ aout) {
  __shared__ __align__(16) unsigned short plds[16 * 32];

  int lane = threadIdx.x;
  int m = lane & 15, hi = lane >> 4;
  int bh = blockIdx.x;          // 0..63
  int qt = blockIdx.y;          // 0..127 (16-row tile index)

  const unsigned short* Qh = qb + (size_t)bh * SEQ * DIM_HEAD;
  const unsigned short* Kh = kb + (size_t)bh * SEQ * DIM_HEAD;
  const unsigned short* Vt = vt + (size_t)bh * DIM_HEAD * SEQ;

  // Q A-fragments (Q is pre-scaled by 1/sqrt(d))
  u16x16 aq[2];
#pragma unroll
  for (int c = 0; c < 2; ++c) {
    const unsigned short* qp =
        Qh + (size_t)(qt * 16 + m) * DIM_HEAD + c * 32 + 8 * hi;
    aq[c] = cat8(*(const u16x8*)qp, *(const u16x8*)(qp + 16));
  }

  f32x8 o[4];
  float mrow[8], lrow[8];
#pragma unroll
  for (int d = 0; d < 4; ++d) o[d] = zero8();
#pragma unroll
  for (int r = 0; r < 8; ++r) { mrow[r] = -3.0e38f; lrow[r] = 0.0f; }

  int ntiles = (qt >> 1) + 1;   // ceil((16*qt+16)/32), causal
  for (int t = 0; t < ntiles; ++t) {
    int j0 = t * 32;

    // ---- S = Q * K^T (16x32 logits = two 16x16 C frags) ----
    f32x8 s[2];
#pragma unroll
    for (int half = 0; half < 2; ++half) {
      s[half] = zero8();
#pragma unroll
      for (int c = 0; c < 2; ++c) {
        const unsigned short* kp =
            Kh + (size_t)(j0 + half * 16 + m) * DIM_HEAD + c * 32 + 8 * hi;
        u16x16 bk = cat8(*(const u16x8*)kp, *(const u16x8*)(kp + 16));
        s[half] = wmma_bf16(aq[c], bk, s[half]);
      }
    }

    // ---- causal mask + row max ----
    float mloc[8];
    int c0 = j0 + m, c1 = j0 + 16 + m;
#pragma unroll
    for (int r = 0; r < 8; ++r) {
      int rr = qt * 16 + r + 8 * hi;
      float a0 = (c0 > rr) ? -3.0e38f : s[0][r];
      float a1 = (c1 > rr) ? -3.0e38f : s[1][r];
      s[0][r] = a0; s[1][r] = a1;
      mloc[r] = fmaxf(a0, a1);
    }
#pragma unroll
    for (int msk = 1; msk < 16; msk <<= 1)
#pragma unroll
      for (int r = 0; r < 8; ++r) mloc[r] = fmaxf(mloc[r], xor_f(mloc[r], msk));

    // ---- online softmax (fp32) ----
    float alpha[8], lsum[8];
#pragma unroll
    for (int r = 0; r < 8; ++r) {
      float mn = fmaxf(mrow[r], mloc[r]);
      alpha[r] = __expf(mrow[r] - mn);
      mrow[r] = mn;
      float p0 = __expf(s[0][r] - mn);
      float p1 = __expf(s[1][r] - mn);
      s[0][r] = p0; s[1][r] = p1;
      lsum[r] = p0 + p1;
    }
#pragma unroll
    for (int msk = 1; msk < 16; msk <<= 1)
#pragma unroll
      for (int r = 0; r < 8; ++r) lsum[r] += xor_f(lsum[r], msk);
#pragma unroll
    for (int r = 0; r < 8; ++r) lrow[r] = lrow[r] * alpha[r] + lsum[r];
#pragma unroll
    for (int d = 0; d < 4; ++d)
#pragma unroll
      for (int r = 0; r < 8; ++r) o[d][r] *= alpha[r];

    // ---- P: C layout -> A layout via LDS (wave-private; dscnt fences) ----
#pragma unroll
    for (int r = 0; r < 8; ++r) {
      plds[(r + 8 * hi) * 32 + m]      = f2bf(s[0][r]);
      plds[(r + 8 * hi) * 32 + 16 + m] = f2bf(s[1][r]);
    }
    asm volatile("s_wait_dscnt 0" ::: "memory");
    const unsigned short* pp = plds + m * 32 + 8 * hi;
    u16x16 ap = cat8(*(const u16x8*)pp, *(const u16x8*)(pp + 16));
    asm volatile("s_wait_dscnt 0" ::: "memory");

    // ---- O += P * V  (V stored transposed [d, n] -> contiguous B frags) ----
#pragma unroll
    for (int d = 0; d < 4; ++d) {
      const unsigned short* vp =
          Vt + (size_t)(d * 16 + m) * SEQ + j0 + 8 * hi;
      u16x16 bv = cat8(*(const u16x8*)vp, *(const u16x8*)(vp + 16));
      o[d] = wmma_bf16(ap, bv, o[d]);
    }
  }

  // ---- normalize + store to [b, n, h*64] bf16 for the output projection ----
  int b = bh >> 4, h = bh & 15;
  float inv[8];
#pragma unroll
  for (int r = 0; r < 8; ++r) inv[r] = 1.0f / lrow[r];
#pragma unroll
  for (int d = 0; d < 4; ++d)
#pragma unroll
    for (int r = 0; r < 8; ++r) {
      int rg = qt * 16 + r + 8 * hi;
      size_t idx = ((size_t)(b * SEQ + rg)) * DMODEL + h * DIM_HEAD + d * 16 + m;
      aout[idx] = f2bf(o[d][r] * inv[r]);
    }
}

// ---------------------------------------------------------------------------
// Kernel 3: out = attn_out @ w_out + b_out (fp32 result)
// grid (1024/64, 8192/128), block 256
// ---------------------------------------------------------------------------
__global__ __launch_bounds__(256) void out_gemm_kernel(
    const unsigned short* __restrict__ ab,
    const unsigned short* __restrict__ woutt,
    const float* __restrict__ bias,
    float* __restrict__ out) {
  __shared__ __align__(16) unsigned short ldsB[2 * 64 * 32];
  int lane = threadIdx.x & 31, wid = threadIdx.x >> 5;
  int m = lane & 15, hi = lane >> 4;
  int row0 = blockIdx.y * 128 + wid * 16;
  int col0 = blockIdx.x * 64;

  f32x8 acc[4];
#pragma unroll
  for (int nn = 0; nn < 4; ++nn) acc[nn] = zero8();

  gemm16x64_lds(ab, woutt, ldsB, row0, col0, acc);

#pragma unroll
  for (int nn = 0; nn < 4; ++nn) {
    int col = col0 + nn * 16 + m;
    float bv = bias[col];
#pragma unroll
    for (int r = 0; r < 8; ++r) {
      int rg = row0 + r + 8 * hi;
      out[(size_t)rg * DMODEL + col] = acc[nn][r] + bv;
    }
  }
}

// ---------------------------------------------------------------------------
extern "C" void kernel_launch(void* const* d_in, const int* in_sizes, int n_in,
                              void* d_out, int out_size, void* d_ws,
                              size_t ws_size, hipStream_t stream) {
  (void)in_sizes; (void)n_in; (void)out_size; (void)ws_size;
  const float* x     = (const float*)d_in[0];
  const float* w_qkv = (const float*)d_in[1];
  const float* w_out = (const float*)d_in[2];
  const float* b_out = (const float*)d_in[3];
  float* out = (float*)d_out;

  char* ws = (char*)d_ws;
  size_t off = 0;
  auto take = [&](size_t bytes) -> void* {
    void* p = ws + off;
    off += (bytes + 255) & ~(size_t)255;
    return p;
  };
  unsigned short* xb    = (unsigned short*)take((size_t)ROWS * DMODEL * 2);
  unsigned short* wqkvt = (unsigned short*)take((size_t)QKV_N * DMODEL * 2);
  unsigned short* woutt = (unsigned short*)take((size_t)DMODEL * DMODEL * 2);
  unsigned short* qb    = (unsigned short*)take((size_t)64 * SEQ * DIM_HEAD * 2);
  unsigned short* kb    = (unsigned short*)take((size_t)64 * SEQ * DIM_HEAD * 2);
  unsigned short* vt    = (unsigned short*)take((size_t)64 * SEQ * DIM_HEAD * 2);
  unsigned short* aout  = (unsigned short*)take((size_t)ROWS * DMODEL * 2);

  cvt_kernel<<<2048, 256, 0, stream>>>(x, xb, ROWS * DMODEL);
  cvt_tr_kernel<DMODEL, QKV_N><<<2048, 256, 0, stream>>>(w_qkv, wqkvt);
  cvt_tr_kernel<DMODEL, DMODEL><<<1024, 256, 0, stream>>>(w_out, woutt);

  qkv_gemm_kernel<<<dim3(QKV_N / 64, ROWS / 128), 256, 0, stream>>>(
      xb, wqkvt, qb, kb, vt);

  attn_kernel<<<dim3(64, SEQ / 16), 32, 0, stream>>>(qb, kb, vt, aout);

  out_gemm_kernel<<<dim3(DMODEL / 64, ROWS / 128), 256, 0, stream>>>(
      aout, woutt, b_out, out);
}
